// GCNII_33973191311764
// MI455X (gfx1250) — matrix operand order repaired and checked by reading, not compiled, not run on hardware
//
#include <hip/hip_runtime.h>
#include <math.h>

#define NN 8192      // nodes
#define NF 512       // feat
#define NH 512       // hidden
#define NC 64        // classes
#define NL 8         // layers

typedef __attribute__((ext_vector_type(16))) __bf16 bf16x16;
typedef __attribute__((ext_vector_type(8)))  float  v8f;
typedef __attribute__((ext_vector_type(8)))  unsigned int u32x8;

__device__ __forceinline__ unsigned short f2bf(float f) {
  unsigned u = __float_as_uint(f);
  u += 0x7fffu + ((u >> 16) & 1u);           // round-to-nearest-even
  return (unsigned short)(u >> 16);
}

union FragU { u32x8 u; bf16x16 b; };

__device__ __forceinline__ bf16x16 load_frag(const unsigned short* p) {
  FragU f;
  f.u = *reinterpret_cast<const u32x8*>(p);  // 32B -> 2x global_load_b128
  return f.b;
}

// ---------------------------------------------------------------------------
// Degree / dinv
// ---------------------------------------------------------------------------
__global__ void k_fill(float* p, float v, int n) {
  int i = blockIdx.x * blockDim.x + threadIdx.x;
  if (i < n) p[i] = v;
}

__global__ void k_colsum(const float* __restrict__ adj, float* __restrict__ deg) {
  int c  = blockIdx.x * 256 + threadIdx.x;
  int r0 = blockIdx.y * 512;
  float s = 0.f;
  for (int r = 0; r < 512; ++r) s += adj[(size_t)(r0 + r) * NN + c];
  atomicAdd(&deg[c], s);
}

__global__ void k_rsqrt(const float* __restrict__ deg, float* __restrict__ dinv, int n) {
  int i = blockIdx.x * blockDim.x + threadIdx.x;
  if (i < n) dinv[i] = rsqrtf(deg[i]);
}

// ---------------------------------------------------------------------------
// Build P = D^-1/2 (A+I) D^-1/2 as bf16, pre-swizzled into WMMA A-fragment
// layout.  A-tile is 16(M) x 32(K); per ISA: lane l<16 -> row l, K pairs
// {0..7,16..23}; lane l>=16 -> row l-16, K pairs {8..15,24..31}.
// Tile storage: [tile][lane(32)][16 bf16] contiguous (1 KB/tile).
// ---------------------------------------------------------------------------
__global__ void k_build_P(const float* __restrict__ adj, const float* __restrict__ dinv,
                          unsigned short* __restrict__ Pp) {
  const int lane = threadIdx.x & 31;
  const int wave = threadIdx.x >> 5;
  const long tile = (long)blockIdx.x * 8 + wave;   // Mt*Kt = 512*256 tiles
  const int Kt = NN / 32;
  const int tm = (int)(tile / Kt), tk = (int)(tile % Kt);
  const int m    = tm * 16 + (lane & 15);
  const int half = lane >> 4;
  const float dm = dinv[m];
  unsigned short* out = Pp + tile * 512 + lane * 16;
  for (int g = 0; g < 2; ++g) {
    const int k0 = tk * 32 + g * 16 + half * 8;
    const float* arow = adj + (size_t)m * NN + k0;
    for (int j = 0; j < 8; ++j) {
      int k = k0 + j;
      float v = arow[j];
      if (k == m) v += 1.0f;
      v *= dm * dinv[k];
      out[g * 8 + j] = f2bf(v);
    }
  }
}

// ---------------------------------------------------------------------------
// Pack fp32 row-major [M][K] -> bf16 A-fragment layout (one thread per 8-run)
// ---------------------------------------------------------------------------
__global__ void k_packA(const float* __restrict__ src, unsigned short* __restrict__ dst,
                        int M, int K) {
  long idx = (long)blockIdx.x * blockDim.x + threadIdx.x;
  long nruns = (long)M * K / 8;
  if (idx >= nruns) return;
  long tile = idx >> 6;                 // 64 runs per tile
  int  rr   = (int)(idx & 63);
  int  lane = rr >> 1;
  int  g    = rr & 1;
  int  Kt   = K >> 5;
  int  tm = (int)(tile / Kt), tk = (int)(tile % Kt);
  int  m    = tm * 16 + (lane & 15);
  int  half = lane >> 4;
  int  k0   = tk * 32 + g * 16 + half * 8;
  const float* s = src + (size_t)m * K + k0;
  unsigned short* d = dst + tile * 512 + (size_t)lane * 16 + g * 8;
  for (int j = 0; j < 8; ++j) d[j] = f2bf(s[j]);
}

// ---------------------------------------------------------------------------
// Pack fp32 row-major [K][N] -> bf16 B-fragment layout.
// B-tile 32(K) x 16(N); lane l<16 -> col l, K={0..7,16..23}; lane l>=16 -> col l-16,
// K={8..15,24..31}.  Tile order (tk*Nt + tn); one thread per lane (16 elems).
// ---------------------------------------------------------------------------
__global__ void k_packB(const float* __restrict__ src, unsigned short* __restrict__ dst,
                        int K, int N) {
  long idx = (long)blockIdx.x * blockDim.x + threadIdx.x;
  long nth = (long)(K >> 5) * (N >> 4) * 32;
  if (idx >= nth) return;
  long tile = idx >> 5;
  int  lane = (int)(idx & 31);
  int  Nt = N >> 4;
  int  tk = (int)(tile / Nt), tn = (int)(tile % Nt);
  int  n  = tn * 16 + (lane & 15);
  int  kb = tk * 32 + (lane >> 4) * 8;
  unsigned short* d = dst + tile * 512 + (size_t)lane * 16;
  for (int s = 0; s < 16; ++s) d[s] = f2bf(src[(size_t)(kb + (s & 7) + (s >> 3) * 16) * N + n]);
}

// ---------------------------------------------------------------------------
// Generic packed-bf16 WMMA GEMM: C[M][N] (fp32) = A_pack x B_pack
// 256 threads = 8 waves (2x4); wave tile 64x64 (4x4 16x16 tiles);
// block tile 128x256; K step 32; 2-stage software-pipelined fragment loads.
// Requires M % 128 == 0, K % 64 == 0, N % 64 == 0 (all call sites satisfy).
// ---------------------------------------------------------------------------
__global__ __launch_bounds__(256) void k_gemm_bf16(
    const unsigned short* __restrict__ A, const unsigned short* __restrict__ B,
    float* __restrict__ C, int M, int N, int K,
    const float* __restrict__ bias, int relu) {
  const int lane = threadIdx.x & 31;
  const int wave = threadIdx.x >> 5;
  const int wr = wave >> 2;                       // 0..1
  const int wc = wave & 3;                        // 0..3
  const int mb = blockIdx.x * 128 + wr * 64;
  const int nb = blockIdx.y * 256 + wc * 64;
  if (nb >= N) return;                            // wave-uniform; EXEC stays all-1s after

  const int Kt = K >> 5;
  const int Nt = N >> 4;
  const int tm0 = mb >> 4;
  const int tn0 = nb >> 4;

  const unsigned short* Ab = A + (size_t)lane * 16;
  const unsigned short* Bb = B + (size_t)lane * 16;

  v8f acc[4][4];
#pragma unroll
  for (int i = 0; i < 4; ++i)
#pragma unroll
    for (int j = 0; j < 4; ++j) acc[i][j] = 0.0f;

  bf16x16 a0[4], b0[4], a1[4], b1[4];
#pragma unroll
  for (int i = 0; i < 4; ++i) a0[i] = load_frag(Ab + ((size_t)(tm0 + i) * Kt + 0) * 512);
#pragma unroll
  for (int j = 0; j < 4; ++j) b0[j] = load_frag(Bb + ((size_t)0 * Nt + (tn0 + j)) * 512);

  for (int kt = 0; kt < Kt; kt += 2) {            // Kt is even at every call site
    const int k1 = kt + 1;
#pragma unroll
    for (int i = 0; i < 4; ++i) a1[i] = load_frag(Ab + ((size_t)(tm0 + i) * Kt + k1) * 512);
#pragma unroll
    for (int j = 0; j < 4; ++j) b1[j] = load_frag(Bb + ((size_t)k1 * Nt + (tn0 + j)) * 512);
#pragma unroll
    for (int i = 0; i < 4; ++i)
#pragma unroll
      for (int j = 0; j < 4; ++j)
        acc[i][j] = __builtin_amdgcn_wmma_f32_16x16x32_bf16(
            false, a0[i], false, b0[j], (short)0, acc[i][j], false, false);

    const int k2 = (kt + 2 < Kt) ? (kt + 2) : kt; // clamped tail prefetch (no branch)
#pragma unroll
    for (int i = 0; i < 4; ++i) a0[i] = load_frag(Ab + ((size_t)(tm0 + i) * Kt + k2) * 512);
#pragma unroll
    for (int j = 0; j < 4; ++j) b0[j] = load_frag(Bb + ((size_t)k2 * Nt + (tn0 + j)) * 512);
#pragma unroll
    for (int i = 0; i < 4; ++i)
#pragma unroll
      for (int j = 0; j < 4; ++j)
        acc[i][j] = __builtin_amdgcn_wmma_f32_16x16x32_bf16(
            false, a1[i], false, b1[j], (short)0, acc[i][j], false, false);
  }

  // Epilogue: C layout — lane holds col = lane%16, rows r + 8*(lane/16)
  const int cl = lane & 15;
  const int rh = (lane >> 4) * 8;
#pragma unroll
  for (int i = 0; i < 4; ++i)
#pragma unroll
    for (int j = 0; j < 4; ++j) {
      const int col = nb + j * 16 + cl;           // < N guaranteed (N%64==0)
      const float bv = bias ? bias[col] : 0.f;
#pragma unroll
      for (int r = 0; r < 8; ++r) {
        const int row = mb + i * 16 + rh + r;
        float v = acc[i][j][r] + bv;
        if (relu) v = fmaxf(v, 0.f);
        C[(size_t)row * N + col] = v;
      }
    }
}

// ---------------------------------------------------------------------------
// out = [relu]( a*X + b*Y )   elementwise, float4-vectorized
// ---------------------------------------------------------------------------
__global__ void k_axpby(const float4* __restrict__ X, const float4* __restrict__ Y,
                        float4* __restrict__ O, float a, float b, int relu, long n4) {
  long i = (long)blockIdx.x * blockDim.x + threadIdx.x;
  if (i >= n4) return;
  float4 x = X[i], y = Y[i], o;
  o.x = a * x.x + b * y.x; o.y = a * x.y + b * y.y;
  o.z = a * x.z + b * y.z; o.w = a * x.w + b * y.w;
  if (relu) {
    o.x = fmaxf(o.x, 0.f); o.y = fmaxf(o.y, 0.f);
    o.z = fmaxf(o.z, 0.f); o.w = fmaxf(o.w, 0.f);
  }
  O[i] = o;
}

// ---------------------------------------------------------------------------
// -log_softmax over rows of [NN][NC]; one wave32 per row, 2 cols/lane
// ---------------------------------------------------------------------------
__global__ void k_logsoftmax(const float* __restrict__ logits, float* __restrict__ out) {
  const int row  = blockIdx.x * 8 + (threadIdx.x >> 5);
  const int lane = threadIdx.x & 31;
  const float* L = logits + (size_t)row * NC;
  float v0 = L[lane], v1 = L[lane + 32];
  float m = fmaxf(v0, v1);
  for (int off = 16; off; off >>= 1) m = fmaxf(m, __shfl_xor(m, off, 32));
  float s = expf(v0 - m) + expf(v1 - m);
  for (int off = 16; off; off >>= 1) s += __shfl_xor(s, off, 32);
  const float c = m + logf(s);
  out[(size_t)row * NC + lane]      = c - v0;
  out[(size_t)row * NC + lane + 32] = c - v1;
}

// ---------------------------------------------------------------------------
extern "C" void kernel_launch(void* const* d_in, const int* in_sizes, int n_in,
                              void* d_out, int out_size, void* d_ws, size_t ws_size,
                              hipStream_t stream) {
  const float* x      = (const float*)d_in[0];
  const float* adj    = (const float*)d_in[1];
  const float* fc1_W  = (const float*)d_in[2];
  const float* fc1_b  = (const float*)d_in[3];
  const float* convWs = (const float*)d_in[4];
  const float* fc2_W  = (const float*)d_in[5];
  const float* fc2_b  = (const float*)d_in[6];
  float* out = (float*)d_out;

  // workspace carve-up (256B aligned)
  char*  ws  = (char*)d_ws;
  size_t off = 0;
  auto alloc = [&](size_t bytes) {
    size_t o = off;
    off = (off + bytes + 255) & ~(size_t)255;
    return o;
  };
  float*          deg   = (float*)(ws + alloc((size_t)NN * 4));
  float*          dinv  = (float*)(ws + alloc((size_t)NN * 4));
  unsigned short* Pp    = (unsigned short*)(ws + alloc((size_t)NN * NN * 2));  // 128 MB
  unsigned short* xA    = (unsigned short*)(ws + alloc((size_t)NN * NF * 2));
  unsigned short* W1B   = (unsigned short*)(ws + alloc((size_t)NF * NH * 2));
  unsigned short* WcB   = (unsigned short*)(ws + alloc((size_t)NL * NH * NH * 2));
  unsigned short* W2B   = (unsigned short*)(ws + alloc((size_t)NH * NC * 2));
  float*          H0    = (float*)(ws + alloc((size_t)NN * NH * 4));
  float*          R     = (float*)(ws + alloc((size_t)NN * NH * 4));
  float*          T     = (float*)(ws + alloc((size_t)NN * NH * 4));
  float*          H     = (float*)(ws + alloc((size_t)NN * NH * 4));
  unsigned short* HB    = (unsigned short*)(ws + alloc((size_t)NN * NH * 2));
  unsigned short* RA    = (unsigned short*)(ws + alloc((size_t)NN * NH * 2));
  float*          logit = (float*)(ws + alloc((size_t)NN * NC * 4));

  const long n4 = (long)NN * NH / 4;

  // 1) dinv = rsqrt(colsum(adj) + 1)
  k_fill<<<NN / 256, 256, 0, stream>>>(deg, 1.0f, NN);
  k_colsum<<<dim3(NN / 256, NN / 512), 256, 0, stream>>>(adj, deg);
  k_rsqrt<<<NN / 256, 256, 0, stream>>>(deg, dinv, NN);

  // 2) P (bf16, A-fragment packed)
  k_build_P<<<(NN / 16) * (NN / 32) / 8, 256, 0, stream>>>(adj, dinv, Pp);

  // 3) pack static operands
  k_packA<<<((long)NN * NF / 8) / 256, 256, 0, stream>>>(x, xA, NN, NF);
  k_packB<<<((NF / 32) * (NH / 16) * 32) / 256, 256, 0, stream>>>(fc1_W, W1B, NF, NH);
  for (int l = 0; l < NL; ++l)
    k_packB<<<((NH / 32) * (NH / 16) * 32) / 256, 256, 0, stream>>>(
        convWs + (size_t)l * NH * NH, WcB + (size_t)l * NH * NH, NH, NH);
  k_packB<<<((NH / 32) * (NC / 16) * 32) / 256, 256, 0, stream>>>(fc2_W, W2B, NH, NC);

  // 4) H0 = relu(x @ fc1_W + b)
  k_gemm_bf16<<<dim3(NN / 128, (NH + 255) / 256), 256, 0, stream>>>(
      xA, W1B, H0, NN, NH, NF, fc1_b, 1);
  k_packB<<<((NN / 32) * (NH / 16) * 32) / 256, 256, 0, stream>>>(H0, HB, NN, NH);

  // 5) GCNII layers
  for (int i = 0; i < NL; ++i) {
    const float beta = logf(0.5f / (float)(i + 1) + 1.0f);
    // T = P @ H
    k_gemm_bf16<<<dim3(NN / 128, (NH + 255) / 256), 256, 0, stream>>>(
        Pp, HB, T, NN, NH, NN, nullptr, 0);
    // R = 0.9*T + 0.1*H0
    k_axpby<<<(int)(n4 / 256), 256, 0, stream>>>(
        (const float4*)T, (const float4*)H0, (float4*)R, 0.9f, 0.1f, 0, n4);
    // T = R @ W_i
    k_packA<<<((long)NN * NH / 8) / 256, 256, 0, stream>>>(R, RA, NN, NH);
    k_gemm_bf16<<<dim3(NN / 128, (NH + 255) / 256), 256, 0, stream>>>(
        RA, WcB + (size_t)i * NH * NH, T, NN, NH, NH, nullptr, 0);
    // H = relu((1-beta)*R + beta*T)
    k_axpby<<<(int)(n4 / 256), 256, 0, stream>>>(
        (const float4*)R, (const float4*)T, (float4*)H, 1.0f - beta, beta, 1, n4);
    if (i < NL - 1)
      k_packB<<<((NN / 32) * (NH / 16) * 32) / 256, 256, 0, stream>>>(H, HB, NN, NH);
  }

  // 6) logits = H @ fc2_W + b2 ; out = -log_softmax
  k_packA<<<((long)NN * NH / 8) / 256, 256, 0, stream>>>(H, RA, NN, NH);
  k_gemm_bf16<<<dim3(NN / 128, (NC + 255) / 256), 256, 0, stream>>>(
      RA, W2B, logit, NN, NC, NH, fc2_b, 0);
  k_logsoftmax<<<NN / 8, 256, 0, stream>>>(logit, out);
}
